// LSTMOpen_46918222742126
// MI455X (gfx1250) — matrix operand order repaired and checked by reading, not compiled, not run on hardware
//
#include <hip/hip_runtime.h>
#include <hip/hip_bf16.h>

#define B_ 64
#define S_ 512
#define I_ 1024
#define H_ 1024
#define G_ 4096   // 4*H

typedef __attribute__((ext_vector_type(16))) _Float16 v16h;
typedef __attribute__((ext_vector_type(4)))  _Float16 v4h;
typedef __attribute__((ext_vector_type(8)))  float    v8f;

union V16H { v16h v; float4 q[2]; };

// A-matrix 16x32 f16 tile per ISA 7.12.2:
//  lanes 0-15  (row M=lane):    elems 0..7 = K kb+0..7,   elems 8..15 = K kb+16..23
//  lanes 16-31 (row M=lane-16): elems 0..7 = K kb+8..15,  elems 8..15 = K kb+24..31
__device__ inline v16h load_a16x32(const _Float16* __restrict__ base, int ld,
                                   int row0, int kb, int lane) {
    int half = lane >> 4, r = lane & 15;
    const _Float16* p = base + (size_t)(row0 + r) * ld + kb + half * 8;
    V16H u;
    u.q[0] = *(const float4*)(p);        // 8 halves
    u.q[1] = *(const float4*)(p + 16);   // 8 halves, K+16
    return u.v;
}

// B-matrix 32x16 f16 tile: lane = N column (W stored [N,K] row-major),
//  lanes 0-15: K kb+0..7, kb+16..23 ; lanes 16-31: K kb+8..15, kb+24..31 (matches A)
__device__ inline v16h load_b16x32(const _Float16* __restrict__ base, int ld,
                                   int col0, int kb, int lane) {
    int half = lane >> 4, n = lane & 15;
    const _Float16* p = base + (size_t)(col0 + n) * ld + kb + half * 8;
    V16H u;
    u.q[0] = *(const float4*)(p);
    u.q[1] = *(const float4*)(p + 16);
    return u.v;
}

__device__ inline float fast_sigmoid(float x) { return 1.f / (1.f + __expf(-x)); }
__device__ inline float fast_tanh(float x)    { return 2.f / (1.f + __expf(-2.f * x)) - 1.f; }

// ---------------------------------------------------------------------------
// Phase-1 GEMM: C = A[M,K] * (B stored [N,K])^T + bias[col]
// Block: 8 waves; wave w: mtile=w&3, ngrp=w>>2. Block tile 64 x 128.
// PF=2 register pipeline over 32-wide K-chunks.
// ---------------------------------------------------------------------------
__global__ __launch_bounds__(256)
void wmma_gemm_bias_kernel(const _Float16* __restrict__ A, int lda,
                           const _Float16* __restrict__ Bm, int ldb,
                           float* __restrict__ C, int ldc,
                           const float* __restrict__ bias, int K) {
    constexpr int NT = 4, PF = 2;
    const int w    = threadIdx.x >> 5;
    const int lane = threadIdx.x & 31;
    const int row0 = blockIdx.y * 64 + (w & 3) * 16;
    const int col0 = blockIdx.x * 128 + (w >> 2) * 64;

    v8f acc[NT];
#pragma unroll
    for (int t = 0; t < NT; ++t) acc[t] = (v8f){0.f,0.f,0.f,0.f,0.f,0.f,0.f,0.f};

    v16h abuf[PF];
    v16h bbuf[PF][NT];
    const int nIter = K >> 5;

#pragma unroll
    for (int s = 0; s < PF; ++s) {
        abuf[s] = load_a16x32(A, lda, row0, s * 32, lane);
#pragma unroll
        for (int t = 0; t < NT; ++t)
            bbuf[s][t] = load_b16x32(Bm, ldb, col0 + t * 16, s * 32, lane);
    }
#pragma unroll PF
    for (int i = 0; i < nIter - PF; ++i) {
        const int slot = i & (PF - 1);
        const int kn = (i + PF) * 32;
#pragma unroll
        for (int t = 0; t < NT; ++t)
            acc[t] = __builtin_amdgcn_wmma_f32_16x16x32_f16(
                false, abuf[slot], false, bbuf[slot][t], (short)0, acc[t], false, false);
        abuf[slot] = load_a16x32(A, lda, row0, kn, lane);
#pragma unroll
        for (int t = 0; t < NT; ++t)
            bbuf[slot][t] = load_b16x32(Bm, ldb, col0 + t * 16, kn, lane);
    }
#pragma unroll PF
    for (int i = nIter - PF; i < nIter; ++i) {
        const int slot = i & (PF - 1);
#pragma unroll
        for (int t = 0; t < NT; ++t)
            acc[t] = __builtin_amdgcn_wmma_f32_16x16x32_f16(
                false, abuf[slot], false, bbuf[slot][t], (short)0, acc[t], false, false);
    }

    const int half = lane >> 4;
    const int nc   = lane & 15;
#pragma unroll
    for (int t = 0; t < NT; ++t) {
        const int col = col0 + t * 16 + nc;
        const float bv = bias[col];
#pragma unroll
        for (int j = 0; j < 8; ++j) {
            const int row = row0 + half * 8 + j;
            C[(size_t)row * ldc + col] = acc[t][j] + bv;
        }
    }
}

// ---------------------------------------------------------------------------
// Fused LSTM step: gates = h @ W_hh^T + x_proj[:,t,:]; then cell update.
// Grid: H/16 = 64 blocks; block bx owns hidden cols j0 = bx*16 .. +15.
// Wave w: mtile = w&3 (rows), gate pair gp = w>>2 -> gates {2gp, 2gp+1}
// (gate g lives at W_hh rows / gate cols g*H + j0). 16 WMMA tiles / block.
// Four 16x16 f32 gate tiles exchanged via LDS, then 256 threads do the
// elementwise cell update on the 64x16 slice (float4 per thread).
// ---------------------------------------------------------------------------
__global__ __launch_bounds__(256)
void lstm_step_kernel(const _Float16* __restrict__ hin,   // [B,H] f16
                      const _Float16* __restrict__ Whh,   // [4H,H] f16
                      const float* __restrict__ xproj_t,  // xproj + t*G, row stride S*G
                      float* __restrict__ c,              // [B,H] f32 (in place)
                      _Float16* __restrict__ hout,        // [B,H] f16 (ping-pong)
                      float* __restrict__ out_h,          // out + t*B*H
                      float* __restrict__ hTp,            // last step only
                      float* __restrict__ cTp) {
    constexpr int PF = 4;
    __shared__ float glds[4][64][16];   // [gate][row][col] = 16 KiB

    const int w    = threadIdx.x >> 5;
    const int lane = threadIdx.x & 31;
    const int mtile = w & 3;
    const int gp    = w >> 2;           // gate pair: {2gp, 2gp+1}
    const int j0    = blockIdx.x * 16;
    const int row0  = mtile * 16;

    v8f acc[2];
    acc[0] = (v8f){0.f,0.f,0.f,0.f,0.f,0.f,0.f,0.f};
    acc[1] = acc[0];

    v16h abuf[PF];
    v16h bbuf[PF][2];
    const int nIter = H_ >> 5;          // 32 chunks

#pragma unroll
    for (int s = 0; s < PF; ++s) {
        abuf[s] = load_a16x32(hin, H_, row0, s * 32, lane);
#pragma unroll
        for (int t = 0; t < 2; ++t)
            bbuf[s][t] = load_b16x32(Whh, H_, (2 * gp + t) * H_ + j0, s * 32, lane);
    }
#pragma unroll PF
    for (int i = 0; i < nIter - PF; ++i) {
        const int slot = i & (PF - 1);
        const int kn = (i + PF) * 32;
#pragma unroll
        for (int t = 0; t < 2; ++t)
            acc[t] = __builtin_amdgcn_wmma_f32_16x16x32_f16(
                false, abuf[slot], false, bbuf[slot][t], (short)0, acc[t], false, false);
        abuf[slot] = load_a16x32(hin, H_, row0, kn, lane);
#pragma unroll
        for (int t = 0; t < 2; ++t)
            bbuf[slot][t] = load_b16x32(Whh, H_, (2 * gp + t) * H_ + j0, kn, lane);
    }
#pragma unroll PF
    for (int i = nIter - PF; i < nIter; ++i) {
        const int slot = i & (PF - 1);
#pragma unroll
        for (int t = 0; t < 2; ++t)
            acc[t] = __builtin_amdgcn_wmma_f32_16x16x32_f16(
                false, abuf[slot], false, bbuf[slot][t], (short)0, acc[t], false, false);
    }

    // Add x_proj addend and stage gate tiles in LDS.
    const int half = lane >> 4;
    const int nc   = lane & 15;
#pragma unroll
    for (int t = 0; t < 2; ++t) {
        const int g = 2 * gp + t;
#pragma unroll
        for (int j = 0; j < 8; ++j) {
            const int row = row0 + half * 8 + j;   // batch row 0..63
            float v = acc[t][j] +
                      xproj_t[(size_t)row * ((size_t)S_ * G_) + (size_t)g * H_ + j0 + nc];
            glds[g][row][nc] = v;
        }
    }
    __syncthreads();

    // Cell update: thread -> (row = tid>>2, 4 cols at (tid&3)*4).
    const int tid  = threadIdx.x;
    const int row  = tid >> 2;
    const int colq = (tid & 3) * 4;
    float4 gi = *(const float4*)&glds[0][row][colq];
    float4 gf = *(const float4*)&glds[1][row][colq];
    float4 gg = *(const float4*)&glds[2][row][colq];
    float4 go = *(const float4*)&glds[3][row][colq];
    const size_t base = (size_t)row * H_ + j0 + colq;
    float4 cv = *(const float4*)(c + base);

    float4 cn, hn;
    cn.x = fast_sigmoid(gf.x) * cv.x + fast_sigmoid(gi.x) * fast_tanh(gg.x);
    cn.y = fast_sigmoid(gf.y) * cv.y + fast_sigmoid(gi.y) * fast_tanh(gg.y);
    cn.z = fast_sigmoid(gf.z) * cv.z + fast_sigmoid(gi.z) * fast_tanh(gg.z);
    cn.w = fast_sigmoid(gf.w) * cv.w + fast_sigmoid(gi.w) * fast_tanh(gg.w);
    hn.x = fast_sigmoid(go.x) * fast_tanh(cn.x);
    hn.y = fast_sigmoid(go.y) * fast_tanh(cn.y);
    hn.z = fast_sigmoid(go.z) * fast_tanh(cn.z);
    hn.w = fast_sigmoid(go.w) * fast_tanh(cn.w);

    *(float4*)(c + base) = cn;
    *(float4*)(out_h + base) = hn;
    v4h h4 = { (_Float16)hn.x, (_Float16)hn.y, (_Float16)hn.z, (_Float16)hn.w };
    *(v4h*)(hout + base) = h4;
    if (hTp) {
        *(float4*)(hTp + base) = hn;
        *(float4*)(cTp + base) = cn;
    }
}

// ---------------------------------------------------------------------------
__global__ void cvt_f32_to_f16_kernel(const float4* __restrict__ in,
                                      v4h* __restrict__ out, int n4) {
    int i = blockIdx.x * blockDim.x + threadIdx.x;
    if (i < n4) {
        float4 v = in[i];
        v4h h = { (_Float16)v.x, (_Float16)v.y, (_Float16)v.z, (_Float16)v.w };
        out[i] = h;
    }
}

__global__ void copy_f32_kernel(const float4* __restrict__ in,
                                float4* __restrict__ out, int n4) {
    int i = blockIdx.x * blockDim.x + threadIdx.x;
    if (i < n4) out[i] = in[i];
}

__global__ void bias_sum_kernel(const float* __restrict__ a,
                                const float* __restrict__ b,
                                float* __restrict__ out, int n) {
    int i = blockIdx.x * blockDim.x + threadIdx.x;
    if (i < n) out[i] = a[i] + b[i];
}

extern "C" void kernel_launch(void* const* d_in, const int* in_sizes, int n_in,
                              void* d_out, int out_size, void* d_ws, size_t ws_size,
                              hipStream_t stream) {
    const float* x   = (const float*)d_in[0];   // [B,S,I]
    const float* h0  = (const float*)d_in[1];   // [B,H]
    const float* c0  = (const float*)d_in[2];   // [B,H]
    const float* Wih = (const float*)d_in[3];   // [4H,I]
    const float* Whh = (const float*)d_in[4];   // [4H,H]
    const float* bih = (const float*)d_in[5];   // [4H]
    const float* bhh = (const float*)d_in[6];   // [4H]
    float* out = (float*)d_out;                 // [S*B*H] + hT[B*H] + cT[B*H]

    char* ws = (char*)d_ws;
    size_t off = 0;
    auto alloc = [&](size_t bytes) -> void* {
        void* p = (void*)(ws + off);
        off += (bytes + 255) & ~(size_t)255;
        return p;
    };

    _Float16* x16    = (_Float16*)alloc((size_t)B_ * S_ * I_ * 2);
    _Float16* Wih16  = (_Float16*)alloc((size_t)G_ * I_ * 2);
    _Float16* Whh16  = (_Float16*)alloc((size_t)G_ * H_ * 2);
    float*    xproj  = (float*)   alloc((size_t)B_ * S_ * G_ * 4);
    _Float16* h16a   = (_Float16*)alloc((size_t)B_ * H_ * 2);
    _Float16* h16b   = (_Float16*)alloc((size_t)B_ * H_ * 2);
    float*    cbuf   = (float*)   alloc((size_t)B_ * H_ * 4);
    float*    bsum   = (float*)   alloc((size_t)G_ * 4);
    (void)ws_size; (void)in_sizes; (void)n_in; (void)out_size;

    // ---- Phase 0: precision conversion + state init (vectorized x4) ----
    {
        int n4 = (B_ * S_ * I_) / 4;
        cvt_f32_to_f16_kernel<<<(n4 + 255) / 256, 256, 0, stream>>>(
            (const float4*)x, (v4h*)x16, n4);
        n4 = (G_ * I_) / 4;
        cvt_f32_to_f16_kernel<<<(n4 + 255) / 256, 256, 0, stream>>>(
            (const float4*)Wih, (v4h*)Wih16, n4);
        n4 = (G_ * H_) / 4;
        cvt_f32_to_f16_kernel<<<(n4 + 255) / 256, 256, 0, stream>>>(
            (const float4*)Whh, (v4h*)Whh16, n4);
        n4 = (B_ * H_) / 4;
        cvt_f32_to_f16_kernel<<<(n4 + 255) / 256, 256, 0, stream>>>(
            (const float4*)h0, (v4h*)h16a, n4);
        copy_f32_kernel<<<(n4 + 255) / 256, 256, 0, stream>>>(
            (const float4*)c0, (float4*)cbuf, n4);
        bias_sum_kernel<<<(G_ + 255) / 256, 256, 0, stream>>>(bih, bhh, bsum, G_);
    }

    // ---- Phase 1: x_proj[B*S, 4H] = x @ W_ih^T + (b_ih + b_hh) ----
    {
        dim3 grid(G_ / 128, (B_ * S_) / 64);
        wmma_gemm_bias_kernel<<<grid, 256, 0, stream>>>(
            x16, I_, Wih16, I_, xproj, G_, bsum, I_);
    }

    // ---- Phase 2: 512 fused steps (GEMM + cell in one kernel) ----
    float* hT = out + (size_t)S_ * B_ * H_;
    float* cT = hT + (size_t)B_ * H_;
    for (int t = 0; t < S_; ++t) {
        const _Float16* hin = (t & 1) ? h16b : h16a;
        _Float16*      hout = (t & 1) ? h16a : h16b;
        bool last = (t == S_ - 1);
        lstm_step_kernel<<<H_ / 16, 256, 0, stream>>>(
            hin, Whh16, xproj + (size_t)t * G_, cbuf, hout,
            out + (size_t)t * B_ * H_,
            last ? hT : nullptr, last ? cT : nullptr);
    }
}
